// QuadrupletMarginMiner_48189533061228
// MI455X (gfx1250) — matrix operand order repaired and checked by reading, not compiled, not run on hardware
//
#include <hip/hip_runtime.h>

typedef __attribute__((ext_vector_type(16))) _Float16 v16h;
typedef __attribute__((ext_vector_type(8)))  float    v8f;
typedef __attribute__((ext_vector_type(2)))  float    v2f;

static constexpr int NB   = 96;   // batch
static constexpr int DF   = 64;   // feature dim
static constexpr int NCLS = 21;   // classes
static constexpr float MARG = 0.3f;

// ---------------------------------------------------------------------------
// Phase 1: one block computes mat[96][96], sames[96][96], diffs[96][96].
// Labels Gram via v_wmma_f32_16x16x32_f16 (exact: labels are 0/1, K=21<=32).
// Cosine Gram via v_wmma_f32_16x16x4_f32 (full f32, K=64 in 16 steps).
// ---------------------------------------------------------------------------
__global__ __launch_bounds__(256) void miner_setup_kernel(
    const float* __restrict__ logits,
    const float* __restrict__ labels,
    const float* __restrict__ feat2,
    float* __restrict__ matW,            // [NB*NB]
    unsigned char* __restrict__ samesW,  // [NB*NB]
    unsigned char* __restrict__ diffsW)  // [NB*NB]
{
    __shared__ _Float16 lab[NB][32];     // labels, f16, zero-padded K 21->32
    __shared__ float    xs[NB][DF];      // normalized logits rows
    __shared__ float    ys[NB][DF];      // normalized feat2 rows

    const int tid = threadIdx.x;

    if (tid < NB) {
        for (int k = 0; k < 32; ++k)
            lab[tid][k] = (k < NCLS) ? (_Float16)labels[tid * NCLS + k]
                                     : (_Float16)0.0f;
        float ss = 0.0f;
        for (int k = 0; k < DF; ++k) { float v = logits[tid * DF + k]; ss += v * v; }
        float inv = 1.0f / fmaxf(sqrtf(ss), 1e-12f);
        for (int k = 0; k < DF; ++k) xs[tid][k] = logits[tid * DF + k] * inv;

        ss = 0.0f;
        for (int k = 0; k < DF; ++k) { float v = feat2[tid * DF + k]; ss += v * v; }
        inv = 1.0f / fmaxf(sqrtf(ss), 1e-12f);
        for (int k = 0; k < DF; ++k) ys[tid][k] = feat2[tid * DF + k] * inv;
    }
    __syncthreads();

    const int wave = tid >> 5;
    const int lane = tid & 31;
    const int lr   = lane & 15;   // row/col within 16-tile
    const int hi   = lane >> 4;   // half-wave select

    // 6x6 = 36 output tiles distributed over 8 waves (wave-uniform loop,
    // EXEC all-ones for every WMMA as required).
    for (int t = wave; t < 36; t += 8) {
        const int ti = t / 6, tj = t % 6;

        // ---- labels Gram: A 16x32 f16, B 32x16 f16 (ISA VGPR layouts) ----
        v16h a, b;
        for (int e = 0; e < 16; ++e) {
            const int ka = (e < 8 ? e : e + 8) + (hi ? 8 : 0);   // A K-index
            const int kb = (e < 8 ? e : e + 8) + (hi ? 8 : 0);   // B K-index
            a[e] = lab[ti * 16 + lr][ka];
            b[e] = lab[tj * 16 + lr][kb];
        }
        v8f gz = {};
        v8f g = __builtin_amdgcn_wmma_f32_16x16x32_f16(
            false, a, false, b, (short)0, gz, false, false);

        // ---- cosine Gram: f32 WMMA 16x16x4, K=64 in 16 steps ----
        v8f acc = {};
        for (int kk = 0; kk < DF; kk += 4) {
            v2f av, bv;
            const int kf = kk + 2 * hi;      // lanes 0-15: K0,K1 ; 16-31: K2,K3
            av.x = xs[ti * 16 + lr][kf];
            av.y = xs[ti * 16 + lr][kf + 1];
            bv.x = ys[tj * 16 + lr][kf];
            bv.y = ys[tj * 16 + lr][kf + 1];
            acc = __builtin_amdgcn_wmma_f32_16x16x4_f32(
                false, av, false, bv, (short)0, acc, false, false);
        }

        // C/D layout: VGPR v -> M = v + 8*hi, N = lane&15
        for (int v = 0; v < 8; ++v) {
            const int r = ti * 16 + v + 8 * hi;
            const int c = tj * 16 + lr;
            const bool share = g[v] > 0.0f;
            samesW[r * NB + c] = (unsigned char)(share && (r != c));
            diffsW[r * NB + c] = (unsigned char)(!share);
            matW[r * NB + c]   = -acc[v];   // mat = -(xn @ yn^T)
        }
    }
}

// ---------------------------------------------------------------------------
// Phase 2: pure store-bandwidth pass. One block per (i, j); 288 threads
// (9 wave32s). n = tid%96 -> coalesced wave stores; k uniform per wave.
// 340 MB of output > 192 MB L2 -> non-temporal stores.
// ---------------------------------------------------------------------------
__global__ __launch_bounds__(288) void miner_emit_kernel(
    const float* __restrict__ matW,
    const unsigned char* __restrict__ samesW,
    const unsigned char* __restrict__ diffsW,
    float* __restrict__ out)
{
    __shared__ float         mrow[NB];
    __shared__ unsigned char srow[NB];
    __shared__ unsigned char drow[NB];

    const int i = blockIdx.y;
    const int j = blockIdx.x;
    const int tid = threadIdx.x;

    if (tid < NB) {
        mrow[tid] = matW[i * NB + tid];
        srow[tid] = samesW[i * NB + tid];
        drow[tid] = diffsW[i * NB + tid];
    }
    __syncthreads();

    const bool  sij = srow[j] != 0;
    const float mij = mrow[j];

    const int n  = tid % NB;     // fastest output dim
    const int k0 = tid / NB;     // 0..2, uniform per wave

    const float dn    = mrow[n];
    const bool  dneg  = drow[n] != 0;
    const bool  violJ = (dn - mij) <= MARG;
    const bool  pre   = sij && dneg;

    float* base = out + (size_t)(i * NB + j) * (NB * NB);

    for (int k = k0; k < NB; k += 3) {
        const bool violK = (dn - mrow[k]) <= MARG;
        const bool v = pre && (srow[k] != 0) && (j < k) && (violJ || violK);
        __builtin_nontemporal_store(v ? 1.0f : 0.0f, base + k * NB + n);
    }
}

// ---------------------------------------------------------------------------
extern "C" void kernel_launch(void* const* d_in, const int* in_sizes, int n_in,
                              void* d_out, int out_size, void* d_ws, size_t ws_size,
                              hipStream_t stream)
{
    (void)in_sizes; (void)n_in; (void)out_size; (void)ws_size;

    const float* logits = (const float*)d_in[0];   // [96,64]
    const float* labels = (const float*)d_in[1];   // [96,21]
    const float* feat2  = (const float*)d_in[2];   // [96,64]
    float* out = (float*)d_out;                    // [96,96,96,96]

    float*         matW   = (float*)d_ws;                       // 96*96 f32
    unsigned char* samesW = (unsigned char*)(matW + NB * NB);   // 96*96 u8
    unsigned char* diffsW = samesW + NB * NB;                   // 96*96 u8

    miner_setup_kernel<<<1, 256, 0, stream>>>(logits, labels, feat2,
                                              matW, samesW, diffsW);
    miner_emit_kernel<<<dim3(NB, NB), 288, 0, stream>>>(matW, samesW, diffsW, out);
}
